// CrossAttention_70609262346703
// MI455X (gfx1250) — matrix operand order repaired and checked by reading, not compiled, not run on hardware
//
#include <hip/hip_runtime.h>
#include <hip/hip_bf16.h>
#include <math.h>

typedef unsigned short bhalf;
typedef unsigned int u32;
typedef __attribute__((ext_vector_type(16))) __bf16 v16bf;
typedef __attribute__((ext_vector_type(8)))  float  v8f;

union Frag { v16bf v; uint4 u[2]; };

__device__ __forceinline__ bhalf f2bf(float f) {
  u32 u = __float_as_uint(f);
  u32 r = 0x7FFFu + ((u >> 16) & 1u);
  return (bhalf)((u + r) >> 16);
}

__device__ __forceinline__ v8f wmma_bf16(v16bf a, v16bf b, v8f c) {
  return __builtin_amdgcn_wmma_f32_16x16x32_bf16(false, a, false, b, (short)0, c, false, false);
}

// ---------------------------------------------------------------------------
// Weight prep: qkv column gather (x-branch keeps q1,k1; ctx-branch keeps q2,v2)
// ---------------------------------------------------------------------------
__global__ __launch_bounds__(256)
void pack_qkv_kernel(const float* __restrict__ qkv_w, const float* __restrict__ qkv_b,
                     bhalf* __restrict__ wxT, bhalf* __restrict__ wcT,
                     float* __restrict__ bqx, float* __restrict__ bqc)
{
  int idx = blockIdx.x * 256 + threadIdx.x;   // 2 * 256 * 256 total
  int e   = idx >> 16;
  int rem = idx & 65535;
  int j   = rem >> 8;     // packed out col
  int k   = rem & 255;    // in dim
  int src;
  if (e == 0) src = (j < 128) ? j          : (256 + (j - 128)); // q1 | k1
  else        src = (j < 128) ? (128 + j)  : (640 + (j - 128)); // q2 | v2
  bhalf val = f2bf(qkv_w[k * 768 + src]);
  if (e == 0) wxT[j * 256 + k] = val; else wcT[j * 256 + k] = val;
  if (k == 0) { if (e == 0) bqx[j] = qkv_b[src]; else bqc[j] = qkv_b[src]; }
}

__global__ __launch_bounds__(256)
void transpose_kernel(const float* __restrict__ W, bhalf* __restrict__ WT, int K, int N)
{
  long idx = (long)blockIdx.x * 256 + threadIdx.x;
  if (idx >= (long)K * N) return;
  long k = idx / N;
  int  n = (int)(idx - k * N);
  WT[(long)n * K + k] = f2bf(W[k * N + n]);
}

// ---------------------------------------------------------------------------
// Workhorse GEMM: C[M=65536, N] = A[M,K] @ W^T[N,K], bf16 WMMA, fused epilogues
// EPI 0: qkv -> scatter bf16 into [win][head][n][hd] layout (roll by `shift`);
//        q columns (c<128) pre-scaled by HD^-0.5 so attention S needs no VALU scale
// EPI 1: fp32 out = acc + bias + aux   (proj+shortcut, fc2+xr)
// EPI 2: bf16 out = gelu(acc + bias)   (fc1)
// ---------------------------------------------------------------------------
#define SA_STRIDE 40

template<int EPI, bool AF32>
__global__ __launch_bounds__(256)
void gemm_kernel(const void* __restrict__ Ap, const bhalf* __restrict__ WT,
                 const float* __restrict__ bias, const float* __restrict__ aux,
                 void* __restrict__ Out, void* __restrict__ Out2,
                 int K, int N, int shift)
{
  __shared__ bhalf sA[128 * SA_STRIDE];
  const int tid  = threadIdx.x;
  const int lane = tid & 31;
  const int wave = tid >> 5;
  const int wm   = wave >> 1;     // 0..3  (M strip)
  const int wn   = wave & 1;      // 0..1  (N strip)
  const long blockM = (long)blockIdx.x * 128;
  const int  blockN = blockIdx.y * 64;

  const int arow  = tid >> 1;         // 0..127
  const int acolh = (tid & 1) << 4;   // 0 or 16

  v8f acc[2][2] = {};

  for (int k0 = 0; k0 < K; k0 += 32) {
    __syncthreads();
    if constexpr (AF32) {
      const float* A = (const float*)Ap;
      const float4* src = (const float4*)(A + (blockM + arow) * (long)K + k0 + acolh);
      float4 f0 = src[0], f1 = src[1], f2 = src[2], f3 = src[3];
      uint4 d0, d1;
      d0.x = (u32)f2bf(f0.x) | ((u32)f2bf(f0.y) << 16);
      d0.y = (u32)f2bf(f0.z) | ((u32)f2bf(f0.w) << 16);
      d0.z = (u32)f2bf(f1.x) | ((u32)f2bf(f1.y) << 16);
      d0.w = (u32)f2bf(f1.z) | ((u32)f2bf(f1.w) << 16);
      d1.x = (u32)f2bf(f2.x) | ((u32)f2bf(f2.y) << 16);
      d1.y = (u32)f2bf(f2.z) | ((u32)f2bf(f2.w) << 16);
      d1.z = (u32)f2bf(f3.x) | ((u32)f2bf(f3.y) << 16);
      d1.w = (u32)f2bf(f3.z) | ((u32)f2bf(f3.w) << 16);
      uint4* dst = (uint4*)&sA[arow * SA_STRIDE + acolh];
      dst[0] = d0; dst[1] = d1;
    } else {
      const bhalf* A = (const bhalf*)Ap;
      const uint4* src = (const uint4*)(A + (blockM + arow) * (long)K + k0 + acolh);
      uint4* dst = (uint4*)&sA[arow * SA_STRIDE + acolh];
      dst[0] = src[0]; dst[1] = src[1];
    }
    if (k0 + 32 < K) {   // -> global_prefetch_b8
      const char* nxt = AF32
        ? (const char*)((const float*)Ap + (blockM + arow) * (long)K + k0 + 32 + acolh)
        : (const char*)((const bhalf*)Ap + (blockM + arow) * (long)K + k0 + 32 + acolh);
      __builtin_prefetch(nxt, 0, 1);
    }
    __syncthreads();

    Frag bfr[2];
    #pragma unroll
    for (int nt = 0; nt < 2; ++nt) {
      int n = blockN + wn * 32 + nt * 16 + (lane & 15);
      const uint4* wp = (const uint4*)(WT + (long)n * K + k0 + ((lane >> 4) << 3));
      bfr[nt].u[0] = wp[0]; bfr[nt].u[1] = wp[2];
    }
    Frag afr[2];
    #pragma unroll
    for (int mt = 0; mt < 2; ++mt) {
      int m  = wm * 32 + mt * 16 + (lane & 15);
      int ko = (lane >> 4) << 3;
      afr[mt].u[0] = *(const uint4*)&sA[m * SA_STRIDE + ko];
      afr[mt].u[1] = *(const uint4*)&sA[m * SA_STRIDE + ko + 16];
    }
    #pragma unroll
    for (int mt = 0; mt < 2; ++mt)
      #pragma unroll
      for (int nt = 0; nt < 2; ++nt)
        acc[mt][nt] = wmma_bf16(afr[mt].v, bfr[nt].v, acc[mt][nt]);
  }

  // epilogue
  #pragma unroll
  for (int mt = 0; mt < 2; ++mt) {
    #pragma unroll
    for (int nt = 0; nt < 2; ++nt) {
      int  c  = blockN + wn * 32 + nt * 16 + (lane & 15);
      long r0 = blockM + wm * 32 + mt * 16 + ((lane >> 4) << 3);
      #pragma unroll
      for (int i = 0; i < 8; ++i) {
        long r = r0 + i;
        float v = acc[mt][nt][i] + bias[c];
        if constexpr (EPI == 0) {
          if (c < 128) v *= 0.25f;          // fold HD^-0.5 into q1/q2
          bhalf bv = f2bf(v);
          int b  = (int)(r >> 14);
          int l_ = (int)(r & 16383);
          int y  = ((l_ >> 7) - shift) & 127;
          int x  = ((l_ & 127) - shift) & 127;
          int n  = ((y & 7) << 3) | (x & 7);
          int wini = (b << 8) | ((y >> 3) << 4) | (x >> 3);
          int cc = c & 127;
          bhalf* outp = (c < 128) ? (bhalf*)Out : (bhalf*)Out2;
          outp[((size_t)(wini * 8 + (cc >> 4)) << 10) | (n << 4) | (cc & 15)] = bv;
        } else if constexpr (EPI == 1) {
          ((float*)Out)[r * 256 + c] = v + aux[r * 256 + c];
        } else {
          ((bhalf*)Out)[r * (long)N + c] =
              f2bf(0.5f * v * (1.0f + erff(v * 0.70710678118f)));
        }
      }
    }
  }
}

// ---------------------------------------------------------------------------
// Window attention: 1 wave per (window, head). Shared S1/S2 logits via WMMA,
// two gated softmax modules in VALU, attn@V via WMMA, fused window_reverse+roll.
// LDS strides padded for 64-bank conflict avoidance; K padding pre-zeroed in sK.
// ---------------------------------------------------------------------------
#define SQ_STRIDE 24   // 48B rows, 16B-aligned frag offsets, bank step 12
#define SK_STRIDE 40   // 80B rows (16 data + 16 zero-pad + 8 pad halves)
#define SV_STRIDE 72   // 144B rows, bank step 36
#define SATT_STRIDE 72
#define SS_STRIDE 65   // 65 floats: softmax lanes (rows 2*lane) spread over banks

__global__ __launch_bounds__(32)
void attn_kernel(const bhalf* __restrict__ q1g, const bhalf* __restrict__ q2g,
                 const bhalf* __restrict__ k1g, const bhalf* __restrict__ v2g,
                 const float* __restrict__ table0, const float* __restrict__ gating0,
                 const float* __restrict__ table1, const float* __restrict__ gating1,
                 bhalf* __restrict__ ybuf)
{
  __shared__ bhalf sQ1[64 * SQ_STRIDE];
  __shared__ bhalf sQ2[64 * SQ_STRIDE];
  __shared__ bhalf sK [64 * SK_STRIDE];
  __shared__ bhalf sVT[16 * SV_STRIDE];
  __shared__ float sS1[64 * SS_STRIDE];
  __shared__ float sS2[64 * SS_STRIDE];
  __shared__ bhalf sAttn[64 * SATT_STRIDE];
  __shared__ float sTab[2][225];
  __shared__ float sInv[64];

  const int lane = threadIdx.x;
  const int blk  = blockIdx.x;
  const int win  = blk >> 3;
  const int head = blk & 7;
  const size_t base = (size_t)blk * 1024;   // (win*8+head)*64*16

  {
    const uint4* a = (const uint4*)(q1g + base);
    const uint4* b = (const uint4*)(q2g + base);
    const uint4* c = (const uint4*)(k1g + base);
    #pragma unroll
    for (int t = 0; t < 4; ++t) {
      int e   = lane + 32 * t;        // uint4 index; row = e>>1, half = e&1
      int row = e >> 1;
      int off = (e & 1) << 3;
      *(uint4*)&sQ1[row * SQ_STRIDE + off] = a[e];
      *(uint4*)&sQ2[row * SQ_STRIDE + off] = b[e];
      *(uint4*)&sK [row * SK_STRIDE + off] = c[e];
    }
    const uint4 z = {0u, 0u, 0u, 0u};
    #pragma unroll
    for (int t = 0; t < 2; ++t) {     // zero K=16..31 pad region of sK
      int row = lane * 2 + t;
      *(uint4*)&sK[row * SK_STRIDE + 16] = z;
      *(uint4*)&sK[row * SK_STRIDE + 24] = z;
    }
    const u32* gv = (const u32*)(v2g + base);
    #pragma unroll
    for (int t = 0; t < 16; ++t) {
      int u = lane * 16 + t;
      u32 w = gv[u];
      int n  = u >> 3;
      int hd = (u & 7) << 1;
      sVT[hd * SV_STRIDE + n]       = (bhalf)(w & 0xffffu);
      sVT[(hd + 1) * SV_STRIDE + n] = (bhalf)(w >> 16);
    }
    for (int e = lane; e < 225; e += 32) {
      sTab[0][e] = table0[e * 8 + head];
      sTab[1][e] = table1[e * 8 + head];
    }
  }
  __syncthreads();

  // S1 = q1 k^T, S2 = q2 k^T  (scale already folded into q; K=16 zero-padded)
  const uint4 z4 = {0u, 0u, 0u, 0u};
  #pragma unroll
  for (int mt = 0; mt < 4; ++mt) {
    Frag a1, a2;
    int m  = mt * 16 + (lane & 15);
    int ko = (lane >> 4) << 3;
    a1.u[0] = *(const uint4*)&sQ1[m * SQ_STRIDE + ko]; a1.u[1] = z4;
    a2.u[0] = *(const uint4*)&sQ2[m * SQ_STRIDE + ko]; a2.u[1] = z4;
    #pragma unroll
    for (int nt = 0; nt < 4; ++nt) {
      Frag b;
      int n  = nt * 16 + (lane & 15);
      int kg = (lane >> 4) << 3;       // u[1] (k 16..31) reads the zeroed pad
      b.u[0] = *(const uint4*)&sK[n * SK_STRIDE + kg];
      b.u[1] = *(const uint4*)&sK[n * SK_STRIDE + kg + 16];
      v8f c1 = {};
      c1 = wmma_bf16(a1.v, b.v, c1);
      v8f c2 = {};
      c2 = wmma_bf16(a2.v, b.v, c2);
      int row0 = mt * 16 + ((lane >> 4) << 3);
      int col  = nt * 16 + (lane & 15);
      #pragma unroll
      for (int i = 0; i < 8; ++i) {
        sS1[(row0 + i) * SS_STRIDE + col] = c1[i];
        sS2[(row0 + i) * SS_STRIDE + col] = c2[i];
      }
    }
  }
  __syncthreads();

  const int winb = win >> 8;
  const int wr   = (win >> 4) & 15;
  const int wc   = win & 15;

  for (int mod = 0; mod < 2; ++mod) {
    float gatev = mod ? gating1[head] : gating0[head];
    float g = 1.0f / (1.0f + __expf(-gatev));
    const float* tab = sTab[mod];

    #pragma unroll
    for (int rr = 0; rr < 2; ++rr) {
      int r  = lane * 2 + rr;
      int yi = r >> 3, xi = r & 7;
      const float* s1r = &sS1[r * SS_STRIDE];
      const float* s2r = &sS2[r * SS_STRIDE];
      float m1 = -1e30f, m2 = -1e30f;
      for (int j = 0; j < 64; ++j) {
        float bb = tab[(yi - (j >> 3) + 7) * 15 + (xi - (j & 7) + 7)];
        m1 = fmaxf(m1, s1r[j] + bb);
        m2 = fmaxf(m2, s2r[j] + bb);
      }
      float s1 = 0.f, s2 = 0.f;
      for (int j = 0; j < 64; ++j) {
        float bb = tab[(yi - (j >> 3) + 7) * 15 + (xi - (j & 7) + 7)];
        s1 += __expf(s1r[j] + bb - m1);
        s2 += __expf(s2r[j] + bb - m2);
      }
      float w1 = (1.0f - g) / s1;
      float w2 = g / s2;
      float sa = 0.f;
      for (int j = 0; j < 64; ++j) {
        float bb = tab[(yi - (j >> 3) + 7) * 15 + (xi - (j & 7) + 7)];
        float a = __expf(s1r[j] + bb - m1) * w1 + __expf(s2r[j] + bb - m2) * w2;
        sa += a;
        sAttn[r * SATT_STRIDE + j] = f2bf(a);
      }
      sInv[r] = 1.0f / sa;   // renorm folded into output scale
    }
    __syncthreads();

    #pragma unroll
    for (int mt = 0; mt < 4; ++mt) {
      v8f c = {};
      #pragma unroll
      for (int ks = 0; ks < 2; ++ks) {
        Frag a, b;
        int m  = mt * 16 + (lane & 15);
        int ko = ks * 32 + ((lane >> 4) << 3);
        a.u[0] = *(const uint4*)&sAttn[m * SATT_STRIDE + ko];
        a.u[1] = *(const uint4*)&sAttn[m * SATT_STRIDE + ko + 16];
        int n  = lane & 15;
        int kg = ks * 32 + ((lane >> 4) << 3);
        b.u[0] = *(const uint4*)&sVT[n * SV_STRIDE + kg];
        b.u[1] = *(const uint4*)&sVT[n * SV_STRIDE + kg + 16];
        c = wmma_bf16(a.v, b.v, c);
      }
      int row0 = mt * 16 + ((lane >> 4) << 3);
      int hd   = lane & 15;
      #pragma unroll
      for (int i = 0; i < 8; ++i) {
        int n = row0 + i;
        float val = c[i] * sInv[n];
        int y = wr * 8 + (n >> 3);
        int x = wc * 8 + (n & 7);
        if (mod) { y = (y + 4) & 127; x = (x + 4) & 127; }   // roll(+SHIFT)
        size_t dst = ((size_t)winb * 16384 + (size_t)y * 128 + x) * 256
                   + mod * 128 + head * 16 + hd;
        ybuf[dst] = f2bf(val);
      }
    }
    __syncthreads();
  }
}

// ---------------------------------------------------------------------------
// LayerNorm over C=256: one wave per row
// ---------------------------------------------------------------------------
__global__ __launch_bounds__(256)
void ln_kernel(const float* __restrict__ xr, const float* __restrict__ gw,
               const float* __restrict__ bw, bhalf* __restrict__ xn)
{
  const int lane = threadIdx.x & 31;
  const int wave = threadIdx.x >> 5;
  const long row = (long)blockIdx.x * 8 + wave;
  const float* p = xr + row * 256;
  float v[8];
  float s = 0.f;
  #pragma unroll
  for (int j = 0; j < 8; ++j) { v[j] = p[lane + 32 * j]; s += v[j]; }
  #pragma unroll
  for (int o = 16; o > 0; o >>= 1) s += __shfl_xor(s, o, 32);
  float mu = s * (1.0f / 256.0f);
  float vs = 0.f;
  #pragma unroll
  for (int j = 0; j < 8; ++j) { float d = v[j] - mu; vs += d * d; }
  #pragma unroll
  for (int o = 16; o > 0; o >>= 1) vs += __shfl_xor(vs, o, 32);
  float rstd = rsqrtf(vs * (1.0f / 256.0f) + 1e-5f);
  #pragma unroll
  for (int j = 0; j < 8; ++j) {
    int c = lane + 32 * j;
    xn[row * 256 + c] = f2bf((v[j] - mu) * rstd * gw[c] + bw[c]);
  }
}

// ---------------------------------------------------------------------------
extern "C" void kernel_launch(void* const* d_in, const int* in_sizes, int n_in,
                              void* d_out, int out_size, void* d_ws, size_t ws_size,
                              hipStream_t stream)
{
  (void)in_sizes; (void)n_in; (void)out_size; (void)ws_size;
  const float* x       = (const float*)d_in[0];
  const float* ctx     = (const float*)d_in[1];
  const float* qkv_w   = (const float*)d_in[2];
  const float* qkv_b   = (const float*)d_in[3];
  const float* proj_w  = (const float*)d_in[4];
  const float* proj_b  = (const float*)d_in[5];
  const float* table0  = (const float*)d_in[6];
  const float* gating0 = (const float*)d_in[7];
  const float* table1  = (const float*)d_in[8];
  const float* gating1 = (const float*)d_in[9];
  const float* norm_g  = (const float*)d_in[10];
  const float* norm_b  = (const float*)d_in[11];
  const float* fc1_w   = (const float*)d_in[12];
  const float* fc1_b   = (const float*)d_in[13];
  const float* fc2_w   = (const float*)d_in[14];
  const float* fc2_b   = (const float*)d_in[15];

  char* ws = (char*)d_ws;
  // weights region (~1.5MB)
  bhalf* wxT   = (bhalf*)ws;
  bhalf* wcT   = wxT   + 256 * 256;
  bhalf* projT = wcT   + 256 * 256;
  bhalf* fc1T  = projT + 256 * 256;
  bhalf* fc2T  = fc1T  + 256 * 1024;
  float* bqx   = (float*)(fc2T + 1024 * 256);
  float* bqc   = bqx + 256;

  // activation arena (phase-overlapped): peak ~226 MB
  const size_t ARENA = 2ull << 20;
  bhalf* q1   = (bhalf*)(ws + ARENA);
  bhalf* k1   = q1 + 8388608ull;
  bhalf* q2   = k1 + 8388608ull;
  bhalf* v2   = q2 + 8388608ull;
  bhalf* ybuf = v2 + 8388608ull;                         // ARENA+64MB, 32MB
  bhalf* xnb  = (bhalf*)(ws + ARENA);                    // reuses q1/k1 (dead)
  bhalf* hb   = (bhalf*)(ws + ARENA + 33554432ull);      // reuses q2/v2/y (dead)
  float* xr   = (float*)(ws + ARENA + 167772160ull);     // live until fc2
  float* outp = (float*)d_out;

  pack_qkv_kernel<<<512, 256, 0, stream>>>(qkv_w, qkv_b, wxT, wcT, bqx, bqc);
  transpose_kernel<<<(256 * 256  + 255) / 256, 256, 0, stream>>>(proj_w, projT, 256, 256);
  transpose_kernel<<<(256 * 1024 + 255) / 256, 256, 0, stream>>>(fc1_w,  fc1T,  256, 1024);
  transpose_kernel<<<(1024 * 256 + 255) / 256, 256, 0, stream>>>(fc2_w,  fc2T,  1024, 256);

  dim3 g4(512, 4), g16(512, 16);
  gemm_kernel<0, true ><<<g4,  256, 0, stream>>>(x,    wxT,   bqx,    nullptr, q1,  k1,     256,  256, 0);
  gemm_kernel<0, true ><<<g4,  256, 0, stream>>>(ctx,  wcT,   bqc,    nullptr, q2,  v2,     256,  256, 4);
  attn_kernel<<<8192, 32, 0, stream>>>(q1, q2, k1, v2, table0, gating0, table1, gating1, ybuf);
  gemm_kernel<1, false><<<g4,  256, 0, stream>>>(ybuf, projT, proj_b, x,       xr,  nullptr, 256,  256, 0);
  ln_kernel<<<8192, 256, 0, stream>>>(xr, norm_g, norm_b, xnb);
  gemm_kernel<2, false><<<g16, 256, 0, stream>>>(xnb,  fc1T,  fc1_b,  nullptr, hb,  nullptr, 256, 1024, 0);
  gemm_kernel<1, false><<<g4,  256, 0, stream>>>(hb,   fc2T,  fc2_b,  xr,      outp, nullptr, 1024, 256, 0);
}
